// Net_31731218383056
// MI455X (gfx1250) — matrix operand order repaired and checked by reading, not compiled, not run on hardware
//
#include <hip/hip_runtime.h>
#include <hip/hip_bf16.h>

// ---------------------------------------------------------------------------
// MI455X (gfx1250, wave32) CNN forward:
//   conv1(f32, fused mask+relu) -> f16, channel-minor layout
//   conv2 as implicit WMMA GEMM with spatial-major K ordering:
//     K-chunk(32) == one 3x3 tap x 32 input channels -> A fragments are
//     two contiguous ds_load_b128 per lane from the LDS-staged image.
//   2x2 maxpool + torch flatten
//   fc1 WMMA GEMM (K=9216) fused bias+relu
//   fc2 WMMA GEMM (N padded 10->16)
//   log_softmax
// ---------------------------------------------------------------------------

typedef _Float16 h16;
typedef __attribute__((ext_vector_type(16))) _Float16 v16h;
typedef __attribute__((ext_vector_type(8)))  _Float16 v8h;
typedef __attribute__((ext_vector_type(8)))  float    v8f;

__device__ __forceinline__ v8f wmma_f32_16x16x32_f16(v16h a, v16h b, v8f c) {
  // (neg_a, A, neg_b, B, c_mod, C, reuse_a, reuse_b)
  return __builtin_amdgcn_wmma_f32_16x16x32_f16(false, a, false, b, (short)0, c,
                                                false, false);
}

// ---------------------------------------------------------------------------
// Kernel 0: weight preparation.
//   conv2: f32 OIHW [oc][ic][ky][kx] -> f16 [oc][(ky*3+kx)*32 + ic]
//          (spatial-major K so each 32-wide WMMA K-chunk = one tap, all ic)
//   fc1  : f32 [128][9216] -> f16 (same order)
//   fc2  : f32 [10][128]   -> f16 [16][128] zero-padded
// ---------------------------------------------------------------------------
__global__ void prep_weights(const float* __restrict__ c2w,
                             const float* __restrict__ f1w,
                             const float* __restrict__ f2w,
                             h16* __restrict__ wg2h,
                             h16* __restrict__ fc1wh,
                             h16* __restrict__ fc2wh) {
  int i = blockIdx.x * blockDim.x + threadIdx.x;   // 0 .. 1179647
  if (i < 64 * 288) {
    int oc = i / 288, t = i - oc * 288;
    int sp = t >> 5, ic = t & 31;                  // sp = ky*3+kx
    wg2h[i] = (h16)c2w[oc * 288 + ic * 9 + sp];
  }
  if (i < 128 * 9216)  fc1wh[i] = (h16)f1w[i];
  if (i < 16 * 128) {
    int n = i >> 7, k = i & 127;
    fc2wh[i] = (n < 10) ? (h16)f2w[n * 128 + k] : (h16)0.f;
  }
}

// ---------------------------------------------------------------------------
// Kernel 1: conv1 (1->32, 3x3) + checkerboard mask + relu, h1 = [b][y][x][ic] f16
// ---------------------------------------------------------------------------
__global__ void conv1_kernel(const float* __restrict__ x,
                             const float* __restrict__ w,
                             const float* __restrict__ bias,
                             h16* __restrict__ h1) {
  int idx = blockIdx.x * blockDim.x + threadIdx.x;   // over 2048*26*26
  if (idx >= 2048 * 676) return;
  int b = idx / 676, r = idx - b * 676;
  int y = r / 26, xx = r - y * 26;
  const float* xp = x + ((size_t)b * 28 + y) * 28 + xx;
  float in[9];
#pragma unroll
  for (int dy = 0; dy < 3; ++dy)
#pragma unroll
    for (int dx = 0; dx < 3; ++dx)
      in[dy * 3 + dx] = xp[dy * 28 + dx];
  int keep = (xx + y) & 1;                 // keep where (i+j) odd
  h16* out = h1 + (size_t)idx * 32;
#pragma unroll
  for (int oc = 0; oc < 32; ++oc) {
    float s = bias[oc];
#pragma unroll
    for (int t = 0; t < 9; ++t) s = fmaf(w[oc * 9 + t], in[t], s);
    s = keep ? fmaxf(s, 0.f) : 0.f;
    out[oc] = (h16)s;
  }
}

// ---------------------------------------------------------------------------
// Kernel 2: conv2 as implicit WMMA GEMM, one block (8 waves) per image.
//   GEMM: [576 x 288] @ [288 x 64]; K-chunk c covers tap (ky,kx)=c, ic=0..31.
//   A fragment = 2x ds_load_b128 from LDS image (channel-minor, contiguous).
//   B fragment = contiguous v16h load from permuted f16 weights [64][288].
//   Epilog: bias + checkerboard mask + relu, h2 = [b][y*24+x][oc] f16.
// ---------------------------------------------------------------------------
__global__ void __launch_bounds__(256) conv2_kernel(
    const h16* __restrict__ h1, const h16* __restrict__ wg2h,
    const float* __restrict__ bias, h16* __restrict__ h2) {
  __shared__ h16 sh[26 * 26 * 32];                 // 43,264 B, [y][x][ic]
  const int b = blockIdx.x;
  const int tid = threadIdx.x;
  // stage whole image into LDS (vectorized 16B chunks; 21632 halves)
  const v8h* src = (const v8h*)(h1 + (size_t)b * 21632);
  v8h* dst = (v8h*)sh;
  for (int i = tid; i < 21632 / 8; i += 256) dst[i] = src[i];
  __syncthreads();

  const int wave = tid >> 5, lane = tid & 31;
  const int g = lane >> 4, n = lane & 15, m = lane & 15;

  for (int mt = wave; mt < 36; mt += 8) {          // 36 M-tiles of 16 rows
    v8f acc[4] = {};
    const int r = mt * 16 + m;                     // this lane's A row
    const int y = r / 24, xx = r - y * 24;
#pragma unroll
    for (int k0 = 0; k0 < 288; k0 += 32) {
      const int sp = k0 >> 5;                      // tap index, compile-time
      const int ky = sp / 3, kx = sp - ky * 3;
      // A fragment: 32 contiguous ic at tap (ky,kx) for this lane's row.
      //   ic = g*16+e for e = 0..15 (contiguous, matches B)
      const h16* p = sh + (size_t)(((y + ky) * 26) + (xx + kx)) * 32 + g * 16;
      v8h lo = *(const v8h*)(p);
      v8h hi = *(const v8h*)(p + 8);
      v16h a;
#pragma unroll
      for (int e = 0; e < 8; ++e) { a[e] = lo[e]; a[e + 8] = hi[e]; }
#pragma unroll
      for (int nt = 0; nt < 4; ++nt) {
        // B fragment: lane n = column, elements = 16 consecutive K
        v16h bf = *(const v16h*)(wg2h + (size_t)(nt * 16 + n) * 288 + k0 + g * 16);
        acc[nt] = wmma_f32_16x16x32_f16(a, bf, acc[nt]);
      }
    }
    // epilog: bias + mask + relu, 16 lanes store 16 contiguous oc
#pragma unroll
    for (int nt = 0; nt < 4; ++nt) {
      int oc = nt * 16 + n;
      float bv = bias[oc];
#pragma unroll
      for (int e = 0; e < 8; ++e) {
        int rr = mt * 16 + g * 8 + e;
        int yy = rr / 24, xc = rr - yy * 24;
        float v = acc[nt][e] + bv;
        v = ((xc + yy) & 1) ? fmaxf(v, 0.f) : 0.f;
        h2[((size_t)b * 576 + rr) * 64 + oc] = (h16)v;
      }
    }
  }
}

// ---------------------------------------------------------------------------
// Kernel 3: 2x2 maxpool + torch flatten -> pooled [2048][c*144+py*12+px] f16
// ---------------------------------------------------------------------------
__global__ void pool_kernel(const h16* __restrict__ h2, h16* __restrict__ pooled) {
  int i = blockIdx.x * blockDim.x + threadIdx.x;   // over 2048*64*12*12
  if (i >= 2048 * 9216) return;
  int b = i / 9216, t = i - b * 9216;
  int c = t / 144, s = t - c * 144;
  int py = s / 12, px = s - py * 12;
  const h16* base = h2 + (size_t)b * 576 * 64 + c;
  float mx = -3.0e38f;
#pragma unroll
  for (int dy = 0; dy < 2; ++dy)
#pragma unroll
    for (int dx = 0; dx < 2; ++dx) {
      float v = (float)base[(size_t)(((2 * py + dy) * 24) + 2 * px + dx) * 64];
      mx = fmaxf(mx, v);
    }
  pooled[(size_t)b * 9216 + t] = (h16)mx;
}

// ---------------------------------------------------------------------------
// Kernel 4: fc1 GEMM [2048,9216]x[9216,128] via WMMA; bias+relu; f16 out.
//   One wave per 16x16 tile; A row-major, B^T row-major -> contiguous loads.
// ---------------------------------------------------------------------------
__global__ void __launch_bounds__(256) fc1_kernel(
    const h16* __restrict__ A, const h16* __restrict__ BT,
    const float* __restrict__ bias, h16* __restrict__ out) {
  const int wave = (blockIdx.x * blockDim.x + threadIdx.x) >> 5;  // 0..1023
  const int lane = threadIdx.x & 31;
  const int tm = wave >> 3, tn = wave & 7;
  const int g = lane >> 4, n = lane & 15, m = lane & 15;
  const h16* arow = A + (size_t)(tm * 16 + m) * 9216 + g * 16;
  const h16* brow = BT + (size_t)(tn * 16 + n) * 9216 + g * 16;
  v8f acc = {};
  for (int k0 = 0; k0 < 9216; k0 += 32) {
    v8h lo = *(const v8h*)(arow + k0);        // K = k0 + g*16 + [0..7]
    v8h hi = *(const v8h*)(arow + k0 + 8);    // K = k0 + g*16 + 8 + [0..7]
    v16h a;
#pragma unroll
    for (int e = 0; e < 8; ++e) { a[e] = lo[e]; a[e + 8] = hi[e]; }
    v16h bf = *(const v16h*)(brow + k0);      // K = k0 + g*16 + [0..15]
    acc = wmma_f32_16x16x32_f16(a, bf, acc);
  }
  const int col = tn * 16 + n;
  const float bv = bias[col];
#pragma unroll
  for (int e = 0; e < 8; ++e) {
    int row = tm * 16 + g * 8 + e;
    out[(size_t)row * 128 + col] = (h16)fmaxf(acc[e] + bv, 0.f);
  }
}

// ---------------------------------------------------------------------------
// Kernel 5: fc2 GEMM [2048,128]x[128,16(pad)] via WMMA; logits f32 [2048][16]
// ---------------------------------------------------------------------------
__global__ void __launch_bounds__(256) fc2_kernel(
    const h16* __restrict__ A, const h16* __restrict__ BT,
    const float* __restrict__ bias, float* __restrict__ out) {
  const int wave = (blockIdx.x * blockDim.x + threadIdx.x) >> 5;  // 0..127
  const int lane = threadIdx.x & 31;
  const int g = lane >> 4, n = lane & 15, m = lane & 15;
  const h16* arow = A + (size_t)(wave * 16 + m) * 128 + g * 16;
  const h16* brow = BT + (size_t)n * 128 + g * 16;
  v8f acc = {};
#pragma unroll
  for (int k0 = 0; k0 < 128; k0 += 32) {
    v8h lo = *(const v8h*)(arow + k0);
    v8h hi = *(const v8h*)(arow + k0 + 8);
    v16h a;
#pragma unroll
    for (int e = 0; e < 8; ++e) { a[e] = lo[e]; a[e + 8] = hi[e]; }
    v16h bf = *(const v16h*)(brow + k0);
    acc = wmma_f32_16x16x32_f16(a, bf, acc);
  }
  const float bv = (n < 10) ? bias[n] : 0.f;
#pragma unroll
  for (int e = 0; e < 8; ++e) {
    int row = wave * 16 + g * 8 + e;
    out[(size_t)row * 16 + n] = acc[e] + bv;
  }
}

// ---------------------------------------------------------------------------
// Kernel 6: log_softmax over 10 logits per row
// ---------------------------------------------------------------------------
__global__ void logsoftmax_kernel(const float* __restrict__ logits,
                                  float* __restrict__ out) {
  int row = blockIdx.x * blockDim.x + threadIdx.x;
  if (row >= 2048) return;
  const float* p = logits + (size_t)row * 16;
  float mx = p[0];
#pragma unroll
  for (int j = 1; j < 10; ++j) mx = fmaxf(mx, p[j]);
  float s = 0.f;
#pragma unroll
  for (int j = 0; j < 10; ++j) s += __expf(p[j] - mx);
  float lse = mx + __logf(s);
#pragma unroll
  for (int j = 0; j < 10; ++j) out[row * 10 + j] = p[j] - lse;
}

// ---------------------------------------------------------------------------
// Launcher
// ---------------------------------------------------------------------------
extern "C" void kernel_launch(void* const* d_in, const int* in_sizes, int n_in,
                              void* d_out, int out_size, void* d_ws, size_t ws_size,
                              hipStream_t stream) {
  const float* x       = (const float*)d_in[0];   // [2048,1,28,28]
  const float* conv1_w = (const float*)d_in[1];   // [32,1,3,3]
  const float* conv1_b = (const float*)d_in[2];   // [32]
  const float* conv2_w = (const float*)d_in[3];   // [64,32,3,3]
  const float* conv2_b = (const float*)d_in[4];   // [64]
  const float* fc1_w   = (const float*)d_in[5];   // [128,9216]
  const float* fc1_b   = (const float*)d_in[6];   // [128]
  const float* fc2_w   = (const float*)d_in[7];   // [10,128]
  const float* fc2_b   = (const float*)d_in[8];   // [10]
  float* outp = (float*)d_out;                    // [2048,10]

  // workspace layout (256B-aligned offsets), ~267 MB total
  char* ws = (char*)d_ws;
  h16*   wg2h   = (h16*)(ws + 0);                 //  64*288   f16 (permuted K)
  h16*   fc1wh  = (h16*)(ws + 36864);             // 128*9216  f16
  h16*   fc2wh  = (h16*)(ws + 2396160);           //  16*128   f16
  h16*   h1     = (h16*)(ws + 2400256);           // 2048*26*26*32 f16
  h16*   h2     = (h16*)(ws + 91004928);          // 2048*576*64   f16
  h16*   pooled = (h16*)(ws + 241999872);         // 2048*9216     f16
  h16*   fc1o   = (h16*)(ws + 279748608);         // 2048*128      f16
  float* fc2o   = (float*)(ws + 280272896);       // 2048*16       f32

  prep_weights<<<4608, 256, 0, stream>>>(conv2_w, fc1_w, fc2_w, wg2h, fc1wh, fc2wh);
  conv1_kernel<<<5408, 256, 0, stream>>>(x, conv1_w, conv1_b, h1);
  conv2_kernel<<<2048, 256, 0, stream>>>(h1, wg2h, conv2_b, h2);
  pool_kernel<<<73728, 256, 0, stream>>>(h2, pooled);
  fc1_kernel<<<128, 256, 0, stream>>>(pooled, fc1wh, fc1_b, fc1o);
  fc2_kernel<<<16, 256, 0, stream>>>(fc1o, fc2wh, fc2_b, fc2o);
  logsoftmax_kernel<<<8, 256, 0, stream>>>(fc2o, outp);
}
